// Net_26671746908502
// MI455X (gfx1250) — matrix operand order repaired and checked by reading, not compiled, not run on hardware
//
#include <hip/hip_runtime.h>
#include <hip/hip_bf16.h>
#include <math.h>

typedef __attribute__((ext_vector_type(16))) __bf16 v16bf;
typedef __attribute__((ext_vector_type(8)))  __bf16 v8bf;
typedef __attribute__((ext_vector_type(8)))  float  v8f;

#define N_NODES 50000
#define N_EDGES 800000
#define E_TOT   (N_EDGES + N_NODES)

__device__ __forceinline__ float lrelu02(float x) { return x > 0.f ? x : 0.2f * x; }

// float atomic max via signed-max (positive) / unsigned-min (negative); init to -inf.
__device__ __forceinline__ void atomicMaxF(float* addr, float val) {
    if (val >= 0.f) atomicMax((int*)addr, __float_as_int(val));
    else            atomicMin((unsigned int*)addr, (unsigned int)__float_as_int(val));
}

// ---------------------------------------------------------------------------
// Pack W[K,N] (f32 row-major) into per-lane bf16 WMMA B-fragments:
//   Bpk[(((kc*ntiles)+tn)*32 + lane)*16 + j] = bf16( W[(kc*32 + (lane>>4)*16 + j)*N + tn*16 + (lane&15)] )
// One thread per packed element; writes fully coalesced.
// ---------------------------------------------------------------------------
__global__ void k_pack_w(const float* __restrict__ W, __bf16* __restrict__ Bpk, int N, int K)
{
    int t = blockIdx.x * blockDim.x + threadIdx.x;
    if (t >= K * N) return;
    const int ntiles = N >> 4;
    int j    = t & 15;
    int lane = (t >> 4) & 31;
    int tn   = (t >> 9) % ntiles;
    int kc   = t / (ntiles << 9);
    int krow = kc * 32 + (lane >> 4) * 16 + j;
    int col  = tn * 16 + (lane & 15);
    Bpk[t] = (__bf16)W[(size_t)krow * N + col];
}

// ---------------------------------------------------------------------------
// C[M,N] = act(A[M,K] @ W[K,N] + bias), W pre-packed as bf16 fragments.
// Grid: M/16 blocks; blockDim = 32*(N/16) (one wave per 16x16 n-tile).
// A slab (16 rows x K) staged in LDS as bf16 once per block.
// Inner loop: 2x ds_load_b128 (A) + 2x global 16B loads (B) + 1x v_wmma.
// ---------------------------------------------------------------------------
#define AMAX_STRIDE 136   // max (K + 8) halves, K <= 128; keeps rows 16B aligned
__global__ void k_wmma_gemm(const float* __restrict__ A, const __bf16* __restrict__ Bpk,
                            const float* __restrict__ bias, float* __restrict__ C,
                            int N, int K, int relu)
{
    __shared__ __align__(16) __bf16 As[16 * AMAX_STRIDE];
    const int stride = K + 8;                 // halves; (K+8)*2 bytes is a multiple of 16
    const int tm     = blockIdx.x;
    const int lane   = threadIdx.x & 31;
    const int tn     = threadIdx.x >> 5;
    const int ntiles = N >> 4;

    // cooperative stage: 16 rows x K floats -> bf16 in LDS (coalesced along K)
    for (int idx = threadIdx.x; idx < 16 * K; idx += blockDim.x) {
        int r = idx / K, c = idx - r * K;
        As[r * stride + c] = (__bf16)A[(size_t)(tm * 16 + r) * K + c];
    }
    __syncthreads();

    const int sel = lane >> 4;                // half-wave select
    const int l15 = lane & 15;

    v8f acc = {};
    for (int k0 = 0; k0 < K; k0 += 32) {
        // A fragment from LDS: halves {k0+sel*16 ..+15}
        const v8bf a_lo = *reinterpret_cast<const v8bf*>(&As[l15 * stride + k0 + sel * 16]);
        const v8bf a_hi = *reinterpret_cast<const v8bf*>(&As[l15 * stride + k0 + sel * 16 + 8]);
        v16bf af = __builtin_shufflevector(a_lo, a_hi,
                                           0,1,2,3,4,5,6,7,8,9,10,11,12,13,14,15);
        // B fragment: contiguous 32B per lane, 1KB coalesced per wave
        const v8bf* bp = reinterpret_cast<const v8bf*>(
            &Bpk[(size_t)((((k0 >> 5) * ntiles + tn) * 32 + lane)) * 16]);
        v16bf bf = __builtin_shufflevector(bp[0], bp[1],
                                           0,1,2,3,4,5,6,7,8,9,10,11,12,13,14,15);

        acc = __builtin_amdgcn_wmma_f32_16x16x32_bf16(false, af, false, bf,
                                                      (short)0, acc, false, false);
    }

    const int col = tn * 16 + l15;
    const float bv = bias ? bias[col] : 0.f;
    #pragma unroll
    for (int v = 0; v < 8; ++v) {
        int r = tm * 16 + sel * 8 + v;        // D layout: VGPR v -> M = v (+8 for upper half-wave)
        float x = acc[v] + bv;
        if (relu) x = fmaxf(x, 0.f);
        C[(size_t)r * N + col] = x;
    }
}

// a_src[n,h] = dot(h[n,h,:], att_src[h,:]);  a_dst likewise
__global__ void k_att(const float* __restrict__ hf, const float* __restrict__ attS,
                      const float* __restrict__ attD, float* __restrict__ aS,
                      float* __restrict__ aD, int H, int F)
{
    int t = blockIdx.x * blockDim.x + threadIdx.x;
    if (t >= N_NODES * H) return;
    int n = t / H, h = t - n * H;
    const float* hp = &hf[(size_t)n * H * F + (size_t)h * F];
    float s0 = 0.f, s1 = 0.f;
    for (int f = 0; f < F; ++f) {
        float v = hp[f];
        s0 += v * attS[h * F + f];
        s1 += v * attD[h * F + f];
    }
    aS[t] = s0; aD[t] = s1;
}

__device__ __forceinline__ void edge_sd(const int* __restrict__ ei, int e, int& s, int& d) {
    if (e < N_EDGES) { s = ei[e]; d = ei[N_EDGES + e]; }
    else             { s = d = e - N_EDGES; }
}

__global__ void k_edge_score_max(const int* __restrict__ ei, const float* __restrict__ aS,
                                 const float* __restrict__ aD, float* __restrict__ score,
                                 float* __restrict__ mB, int H)
{
    long long t = (long long)blockIdx.x * blockDim.x + threadIdx.x;
    if (t >= (long long)E_TOT * H) return;
    int e = (int)(t / H), h = (int)(t - (long long)e * H);
    int s, d; edge_sd(ei, e, s, d);
    float sc = lrelu02(aS[s * H + h] + aD[d * H + h]);
    score[t] = sc;
    atomicMaxF(&mB[d * H + h], sc);
}

__global__ void k_edge_exp_sum(const int* __restrict__ ei, float* __restrict__ score,
                               const float* __restrict__ mB, float* __restrict__ sB, int H)
{
    long long t = (long long)blockIdx.x * blockDim.x + threadIdx.x;
    if (t >= (long long)E_TOT * H) return;
    int e = (int)(t / H), h = (int)(t - (long long)e * H);
    int s, d; edge_sd(ei, e, s, d);
    float ex = expf(score[t] - mB[d * H + h]);
    score[t] = ex;
    atomicAdd(&sB[d * H + h], ex);
}

__global__ void k_edge_aggregate(const int* __restrict__ ei, const float* __restrict__ score,
                                 const float* __restrict__ sB, const float* __restrict__ hf,
                                 float* __restrict__ outb, int H, int F)
{
    long long t = (long long)blockIdx.x * blockDim.x + threadIdx.x;
    const int HF = H * F;
    if (t >= (long long)E_TOT * HF) return;
    int e = (int)(t / HF);
    int c = (int)(t - (long long)e * HF);
    int h = c / F;
    int s, d; edge_sd(ei, e, s, d);
    float alpha = score[(size_t)e * H + h] / (sB[d * H + h] + 1e-16f);
    atomicAdd(&outb[(size_t)d * HF + c], alpha * hf[(size_t)s * HF + c]);
}

__global__ void k_fill(float* __restrict__ p, float v, long long n)
{
    long long t = (long long)blockIdx.x * blockDim.x + threadIdx.x;
    if (t < n) p[t] = v;
}

__global__ void k_bias_act(float* __restrict__ outb, const float* __restrict__ bias,
                           int C, int relu)
{
    long long t = (long long)blockIdx.x * blockDim.x + threadIdx.x;
    if (t >= (long long)N_NODES * C) return;
    int c = (int)(t % C);
    float x = outb[t] + bias[c];
    if (relu) x = fmaxf(x, 0.f);
    outb[t] = x;
}

extern "C" void kernel_launch(void* const* d_in, const int* in_sizes, int n_in,
                              void* d_out, int out_size, void* d_ws, size_t ws_size,
                              hipStream_t stream)
{
    const float* x   = (const float*)d_in[0];
    const int*   ei  = (const int*)d_in[1];
    const float *W1  = (const float*)d_in[2],  *b1  = (const float*)d_in[3];
    const float *W2  = (const float*)d_in[4],  *b2  = (const float*)d_in[5];
    const float *Wg1 = (const float*)d_in[6],  *as1 = (const float*)d_in[7];
    const float *ad1 = (const float*)d_in[8],  *bg1 = (const float*)d_in[9];
    const float *Wg2 = (const float*)d_in[10], *as2 = (const float*)d_in[11];
    const float *ad2 = (const float*)d_in[12], *bg2 = (const float*)d_in[13];
    const float *Wg3 = (const float*)d_in[14], *as3 = (const float*)d_in[15];
    const float *ad3 = (const float*)d_in[16], *bg3 = (const float*)d_in[17];
    float* out = (float*)d_out;

    float* ws = (float*)d_ws;
    const size_t NB = (size_t)N_NODES * 128;
    float* bufA = ws;                 // 50000x128
    float* bufB = ws + NB;            // 50000x128
    float* bufC = ws + 2 * NB;        // 50000x128
    float* aS   = ws + 3 * NB;        // 50000x2
    float* aD   = aS + (size_t)N_NODES * 2;
    float* mB   = aD + (size_t)N_NODES * 2;
    float* sB   = mB + (size_t)N_NODES * 2;
    float* ex   = sB + (size_t)N_NODES * 2;        // E_TOT x 2
    __bf16* pw  = (__bf16*)(ex + (size_t)E_TOT * 2);
    __bf16* pw1  = pw;                 // 128x32
    __bf16* pw2  = pw1 + 128 * 32;     // 32x64
    __bf16* pwg1 = pw2 + 32 * 64;      // 64x64
    __bf16* pwg2 = pwg1 + 64 * 64;     // 64x128
    __bf16* pwg3 = pwg2 + 64 * 128;    // 128x128

    auto cdiv = [](long long a, long long b) { return (int)((a + b - 1) / b); };

    auto pack = [&](const float* W, __bf16* Bpk, int N, int K) {
        k_pack_w<<<cdiv((long long)K * N, 256), 256, 0, stream>>>(W, Bpk, N, K);
    };
    auto gemm = [&](const float* A, const __bf16* Bpk, const float* bias, float* C,
                    int N, int K, int relu) {
        k_wmma_gemm<<<N_NODES / 16, 32 * (N / 16), 0, stream>>>(A, Bpk, bias, C, N, K, relu);
    };

    auto gat = [&](const float* xin, int Kin, const __bf16* Wpk, const float* attS,
                   const float* attD, const float* bg, float* hf, float* outb,
                   int H, int F, int relu) {
        const int C = H * F;
        gemm(xin, Wpk, nullptr, hf, C, Kin, 0);                        // h = x @ W (no bias yet)
        long long nh = (long long)N_NODES * H;
        k_att<<<cdiv(nh, 256), 256, 0, stream>>>(hf, attS, attD, aS, aD, H, F);
        k_fill<<<cdiv(nh, 256), 256, 0, stream>>>(mB, -__builtin_huge_valf(), nh);
        k_fill<<<cdiv(nh, 256), 256, 0, stream>>>(sB, 0.f, nh);
        long long nc = (long long)N_NODES * C;
        k_fill<<<cdiv(nc, 256), 256, 0, stream>>>(outb, 0.f, nc);
        long long eh = (long long)E_TOT * H;
        k_edge_score_max<<<cdiv(eh, 256), 256, 0, stream>>>(ei, aS, aD, ex, mB, H);
        k_edge_exp_sum<<<cdiv(eh, 256), 256, 0, stream>>>(ei, ex, mB, sB, H);
        long long ehf = (long long)E_TOT * C;
        k_edge_aggregate<<<cdiv(ehf, 256), 256, 0, stream>>>(ei, ex, sB, hf, outb, H, F);
        k_bias_act<<<cdiv(nc, 256), 256, 0, stream>>>(outb, bg, C, relu);
    };

    // pre-pack all weights into bf16 WMMA fragment layout (tiny, L2-resident)
    pack(W1,  pw1,  32,  128);
    pack(W2,  pw2,  64,  32);
    pack(Wg1, pwg1, 64,  64);
    pack(Wg2, pwg2, 128, 64);
    pack(Wg3, pwg3, 128, 128);

    // MLP stem (bias+ReLU fused in GEMM epilogue)
    gemm(x,    pw1, b1, bufB, 32, 128, 1);    // h1: 50000x32  (in bufB)
    gemm(bufB, pw2, b2, bufA, 64, 32,  1);    // h2: 50000x64  (in bufA)

    // GAT layers (hf scratch = bufB each time; outputs ping-pong)
    gat(bufA, 64,  pwg1, as1, ad1, bg1, bufB, bufC, 2, 32,  1);   // -> 50000x64  in bufC
    gat(bufC, 64,  pwg2, as2, ad2, bg2, bufB, bufA, 2, 64,  1);   // -> 50000x128 in bufA
    gat(bufA, 128, pwg3, as3, ad3, bg3, bufB, out,  1, 128, 0);   // -> 50000x128 in d_out
}
